// GaussBAE_29798483100092
// MI455X (gfx1250) — matrix operand order repaired and checked by reading, not compiled, not run on hardware
//
#include <hip/hip_runtime.h>

// GaussBAE E-step for MI455X (gfx1250, wave32).
// Stage 1: C = 2*X@W - (2*b@W + scl) via v_wmma_f32_16x16x32_bf16, X tiles
//          staged into LDS by the Tensor Data Mover (double-buffered).
// Stage 2: sequential Gibbs sweep, one wave32, lane-per-latent layout.

#define NN 8192
#define DD 2048
#define RR 32
#define TEMP_INV 10.0f
#define BETA_F   1.0f

typedef __attribute__((ext_vector_type(16))) __bf16          v16bf;
typedef __attribute__((ext_vector_type(8)))  float           v8f;
typedef __attribute__((ext_vector_type(8)))  unsigned short  us8;
typedef __attribute__((ext_vector_type(4)))  unsigned int    u32x4;
typedef __attribute__((ext_vector_type(8)))  int             i32x8;
typedef __attribute__((ext_vector_type(4)))  int             i32x4;

union Frag16 { us8 u[2]; v16bf v; };

#if __has_builtin(__builtin_amdgcn_tensor_load_to_lds) && \
    __has_builtin(__builtin_amdgcn_s_wait_tensorcnt)
#define USE_TDM 1
#else
#define USE_TDM 0
#endif

__device__ __forceinline__ unsigned short f2bf(float f) {
  __bf16 h = (__bf16)f;                         // native v_cvt on gfx1250
  return __builtin_bit_cast(unsigned short, h);
}

__device__ __forceinline__ us8 pack8(float4 x, float4 y) {
  us8 r = { f2bf(x.x), f2bf(x.y), f2bf(x.z), f2bf(x.w),
            f2bf(y.x), f2bf(y.y), f2bf(y.z), f2bf(y.w) };
  return r;
}

__device__ __forceinline__ float sigmoidf(float x) {
  return 1.0f / (1.0f + __expf(-x));
}

// ---------------- threefry2x32 (20 rounds), JAX's RNG family ----------------
__device__ __forceinline__ void tf_mix(unsigned& x0, unsigned& x1, int r) {
  x0 += x1; x1 = (x1 << r) | (x1 >> (32 - r)); x1 ^= x0;
}
__device__ __forceinline__ void threefry2x32(unsigned k0, unsigned k1,
                                             unsigned c0, unsigned c1,
                                             unsigned& o0, unsigned& o1) {
  unsigned k2 = k0 ^ k1 ^ 0x1BD11BDAu;
  unsigned x0 = c0 + k0, x1 = c1 + k1;
  tf_mix(x0, x1, 13); tf_mix(x0, x1, 15); tf_mix(x0, x1, 26); tf_mix(x0, x1, 6);
  x0 += k1; x1 += k2 + 1u;
  tf_mix(x0, x1, 17); tf_mix(x0, x1, 29); tf_mix(x0, x1, 16); tf_mix(x0, x1, 24);
  x0 += k2; x1 += k0 + 2u;
  tf_mix(x0, x1, 13); tf_mix(x0, x1, 15); tf_mix(x0, x1, 26); tf_mix(x0, x1, 6);
  x0 += k0; x1 += k1 + 3u;
  tf_mix(x0, x1, 17); tf_mix(x0, x1, 29); tf_mix(x0, x1, 16); tf_mix(x0, x1, 24);
  x0 += k1; x1 += k2 + 4u;
  tf_mix(x0, x1, 13); tf_mix(x0, x1, 15); tf_mix(x0, x1, 26); tf_mix(x0, x1, 6);
  x0 += k2; x1 += k0 + 5u;
  o0 = x0; o1 = x1;
}
__device__ __forceinline__ float u01(unsigned x) { return (float)(x >> 8) * 0x1p-24f; }

// ---------------- TDM: 2D fp32 tile load (wave-uniform operands) ------------
#if USE_TDM
__device__ __forceinline__ void tdm_load_tile32x64(unsigned lds_off,
                                                   unsigned long long gaddr) {
  u32x4 g0;
  g0[0] = 1u;                                            // count=1, user, load
  g0[1] = lds_off;                                       // lds_addr (bytes)
  g0[2] = (unsigned)(gaddr & 0xFFFFFFFFu);               // global_addr[31:0]
  g0[3] = (unsigned)((gaddr >> 32) & 0x01FFFFFFu) | (2u << 30); // [56:32] | type=2
  i32x8 g1;
  g1[0] = (int)(2u << 16);                 // wg_mask=0, data_size=2 (4 bytes)
  g1[1] = (int)((unsigned)(DD & 0xFFFF) << 16);          // tensor_dim0[15:0]
  g1[2] = (int)(((unsigned)DD >> 16) | ((unsigned)(NN & 0xFFFF) << 16)); // d0 hi | d1 lo
  g1[3] = (int)(((unsigned)NN >> 16) | (32u << 16));     // d1 hi | tile_dim0=32
  g1[4] = (int)(64u);                                    // tile_dim1=64, tile_dim2=0
  g1[5] = (int)(DD);                                     // tensor_dim0_stride lo
  g1[6] = 0;                                             // stride hi | d1_stride lo
  g1[7] = 0;
  i32x4 z4 = {};
#if defined(__clang_major__) && (__clang_major__ >= 23)
  i32x8 z8 = {};
  __builtin_amdgcn_tensor_load_to_lds(g0, g1, z4, z4, z8, 0);
#else
  __builtin_amdgcn_tensor_load_to_lds(g0, g1, z4, z4, 0);
#endif
}
#endif

// ---------------- prep: W (D,R) fp32 row-major -> Wt (R,D) bf16 -------------
__global__ __launch_bounds__(256) void wprep_kernel(const float* __restrict__ W,
                                                    unsigned short* __restrict__ Wt) {
  int t = blockIdx.x * 256 + threadIdx.x;
  int k = t >> 5, n = t & 31;
  Wt[(size_t)n * DD + k] = f2bf(W[(size_t)k * RR + n]);
}

// ---------------- prep: adj[n] = 2*(b@W)[n] + scl ---------------------------
__global__ __launch_bounds__(32) void adj_kernel(const float* __restrict__ W,
                                                 const float* __restrict__ b,
                                                 const float* __restrict__ scl,
                                                 float* __restrict__ adj) {
  int n = threadIdx.x;
  float s = 0.0f;
  for (int d = 0; d < DD; ++d) s += b[d] * W[d * RR + n];
  adj[n] = 2.0f * s + scl[0];
}

// ---------------- GEMM: Cout = 2*X@W - adj, via bf16 WMMA -------------------
// 128 threads = 4 waves; WG covers 64 rows x 32 cols of C.
// X fp32 tiles (64x32) DMA'd into LDS by TDM (double-buffered); waves convert
// to bf16 on read and issue v_wmma_f32_16x16x32_bf16 (fp32 accumulate).
__global__ __launch_bounds__(128) void gemm_kernel(const float* __restrict__ X,
                                                   const unsigned short* __restrict__ Wt,
                                                   const float* __restrict__ adj,
                                                   float* __restrict__ Cout) {
  __shared__ float xt[2][64 * 32];          // 2 x 8 KiB fp32 tiles

  const int tid  = threadIdx.x;
  const int wave = tid >> 5;
  const int lane = tid & 31;
  const int row0 = blockIdx.x * 64;

  // WMMA lane mapping (ISA 7.12.2): A 16x32 bf16, C/D 16x16 f32
  const int nlo   = lane & 15;              // A row within slab / B,D column
  const int hi    = lane >> 4;
  const int akoff = hi * 8;                 // A: K {0..7}|{8..15}, plus +16 block
  const int bkoff = hi * 8;                 // B: K {0..7}|{8..15}, plus +16 block
  const int arow  = wave * 16 + nlo;
  const unsigned short* wt0 = Wt + (size_t)nlo        * DD + bkoff;
  const unsigned short* wt1 = Wt + (size_t)(16 + nlo) * DD + bkoff;

  v8f c0 = {}, c1 = {};

#if USE_TDM
  const unsigned long long xbase = (unsigned long long)(X + (size_t)row0 * DD);
  const unsigned lds0 = (unsigned)(unsigned long long)&xt[0][0];
  const unsigned lds1 = (unsigned)(unsigned long long)&xt[1][0];
  if (wave == 0) tdm_load_tile32x64(lds0, xbase);        // kb = 0
#endif

  int cur = 0;
  for (int kb = 0; kb < DD; kb += 32) {
#if USE_TDM
    if (wave == 0) {
      if (kb + 32 < DD) {
        tdm_load_tile32x64(cur ? lds0 : lds1,
                           xbase + (unsigned long long)(kb + 32) * 4u);
        __builtin_amdgcn_s_wait_tensorcnt(1);   // current tile landed
      } else {
        __builtin_amdgcn_s_wait_tensorcnt(0);
      }
    }
    __syncthreads();
#else
    __syncthreads();
    {   // cooperative fallback: 128 threads x 16 floats, coalesced 128B/row
      const int lrow = tid >> 1, lhalf = tid & 1;
      const float4* src = (const float4*)(X + (size_t)(row0 + lrow) * DD + kb + lhalf * 16);
      float4* dst = (float4*)&xt[cur][lrow * 32 + lhalf * 16];
      dst[0] = src[0]; dst[1] = src[1]; dst[2] = src[2]; dst[3] = src[3];
      if (kb + 32 < DD) __builtin_prefetch(src + 4, 0, 0);
    }
    __syncthreads();
#endif

    const float* ap = &xt[cur][arow * 32 + akoff];
    float4 a0 = ((const float4*)ap)[0];
    float4 a1 = ((const float4*)ap)[1];
    float4 a2 = ((const float4*)(ap + 16))[0];
    float4 a3 = ((const float4*)(ap + 16))[1];

    Frag16 a, b0, b1;
    a.u[0]  = pack8(a0, a1);                  // K =  akoff .. akoff+7
    a.u[1]  = pack8(a2, a3);                  // K = 16+akoff .. 23+akoff
    b0.u[0] = ((const us8*)(wt0 + kb))[0];
    b0.u[1] = ((const us8*)(wt0 + kb + 16))[0];
    b1.u[0] = ((const us8*)(wt1 + kb))[0];
    b1.u[1] = ((const us8*)(wt1 + kb + 16))[0];

    c0 = __builtin_amdgcn_wmma_f32_16x16x32_bf16(false, a.v, false, b0.v,
                                                 (short)0, c0, false, false);
    c1 = __builtin_amdgcn_wmma_f32_16x16x32_bf16(false, a.v, false, b1.v,
                                                 (short)0, c1, false, false);
    __syncthreads();
    cur ^= 1;
  }

  // D layout: VGPR j -> M = j (lanes 0-15) or 8+j (lanes 16-31), N = lane&15
  const int mbase = row0 + wave * 16 + hi * 8;
  const float a0j = adj[nlo];
  const float a1j = adj[16 + nlo];
#pragma unroll
  for (int jj = 0; jj < 8; ++jj) {
    const int r = mbase + jj;
    Cout[r * RR + nlo]      = 2.0f * c0[jj] - a0j;
    Cout[r * RR + 16 + nlo] = 2.0f * c1[jj] - a1j;
  }
}

// ---------------- sequential Gibbs sweep: one wave32 ------------------------
// Lane j owns column j: StS[:,j] in 32 VGPRs, St1[j], s[j].
__global__ __launch_bounds__(32) void gibbs_kernel(const float* __restrict__ C,
                                                   float* __restrict__ ES,
                                                   unsigned* __restrict__ Sbits) {
  const int j = threadIdx.x;
  __shared__ int Qlds[RR * RR];

  int sts[RR];
#pragma unroll
  for (int i = 0; i < RR; ++i) sts[i] = 0;
  int st1 = 0;

  // ---- init: S ~ Bernoulli(sigmoid(C/temp)); StS = S^T S; St1 = S^T 1
  for (int r = 0; r < NN; ++r) {
    float p = sigmoidf(C[r * RR + j] * TEMP_INV);
    unsigned u0, u1;
    threefry2x32(0x9E3779B9u, 0x42u, (unsigned)r, (unsigned)j, u0, u1);
    int sj = (u01(u0) < p) ? 1 : 0;
    unsigned m = (unsigned)__ballot(sj);
    if (j == 0) Sbits[r] = m;
    if (sj) {
#pragma unroll
      for (int i = 0; i < RR; ++i) sts[i] += (int)((m >> i) & 1u);
    }
    st1 += sj;
  }

  const float nm1 = (float)(NN - 1);

  for (int t = 0; t < NN; ++t) {
    const int it = (int)(((unsigned)t * 0x9E3779B1u) & (NN - 1));  // row perm
    const unsigned m = Sbits[it];
    int sj = (int)((m >> j) & 1u);

    // rank-1 downdate
#pragma unroll
    for (int i = 0; i < RR; ++i) sts[i] -= sj & (int)((m >> i) & 1u);
    st1 -= sj;

    const float st1_j = (float)st1;
    int p_j = 0;
#pragma unroll
    for (int i = 0; i < RR; ++i) {
      float st1_i = __shfl(st1_j, i, RR);
      float d1 = (float)sts[i];
      float d2 = st1_j - d1;
      float d3 = st1_i - d1;
      float d4 = nm1 - st1_j - st1_i + d1;
      float mn = fminf(fminf(d1, d2), fminf(d3, d4));
      int b1 = (d1 == mn), b2 = (d2 == mn), b3 = (d3 == mn), b4 = (d4 == mn);
      int uniq = ((b1 + b2 + b3 + b4) == 1) ? -1 : 0;   // zero out ties
      b1 &= uniq; b2 &= uniq; b3 &= uniq; b4 &= uniq;
      Qlds[i * RR + j] = b1 - b2 - b3 + b4;             // Q[i][j]
      p_j += b2 - b4;                                   // p[j] = sum_i(b2-b4)
    }
    __syncthreads();

    const float crow_j = C[it * RR + j];
    unsigned pr0, pr1;
    threefry2x32(0xC0FFEEu, 0x7u, (unsigned)it, 0u, pr0, pr1);
    const int imul = (int)(pr0 | 1u);                   // fresh inner permutation
    const int ioff = (int)(pr1 & 31u);
    const float p_jf = (float)p_j;

    for (int t2 = 0; t2 < RR; ++t2) {
      int i = (t2 * imul + ioff) & (RR - 1);
      float part = (float)(Qlds[i * RR + j] * sj);      // Q[i][j] * s[j]
#pragma unroll
      for (int off = 16; off > 0; off >>= 1) part += __shfl_xor(part, off, RR);
      float inhib = part + __shfl(p_jf, i, RR);
      float ci    = __shfl(crow_j, i, RR);
      float e     = sigmoidf((ci - BETA_F * inhib) * TEMP_INV);
      float prob  = sigmoidf(e);                        // reference double-sigmoid
      unsigned g0, g1;
      threefry2x32(0xD1CEu, 0x5EEDu, (unsigned)(it * RR + t2), 0u, g0, g1);
      int samp = (u01(g0) < prob) ? 1 : 0;
      if (j == i) {
        sj = samp;
        ES[it * RR + i] = e;
      }
    }

    unsigned m2 = (unsigned)__ballot(sj);
    if (j == 0) Sbits[it] = m2;
#pragma unroll
    for (int i = 0; i < RR; ++i) sts[i] += sj & (int)((m2 >> i) & 1u);
    st1 += sj;
    __syncthreads();
  }
}

extern "C" void kernel_launch(void* const* d_in, const int* in_sizes, int n_in,
                              void* d_out, int out_size, void* d_ws, size_t ws_size,
                              hipStream_t stream) {
  (void)in_sizes; (void)n_in; (void)out_size; (void)ws_size;
  const float* X   = (const float*)d_in[0];
  const float* W   = (const float*)d_in[1];
  const float* b   = (const float*)d_in[2];
  const float* scl = (const float*)d_in[3];
  float* ES = (float*)d_out;

  char* ws = (char*)d_ws;
  float*          C     = (float*)ws;                                  // 1 MiB
  unsigned short* Wt    = (unsigned short*)(ws + (1u << 20));          // 128 KiB
  float*          adj   = (float*)(ws + (1u << 20) + (128u << 10));    // 128 B
  unsigned*       Sbits = (unsigned*)(ws + (1u << 20) + (128u << 10) + 128); // 32 KiB

  wprep_kernel<<<(DD * RR) / 256, 256, 0, stream>>>(W, Wt);
  adj_kernel<<<1, 32, 0, stream>>>(W, b, scl, adj);
  gemm_kernel<<<NN / 64, 128, 0, stream>>>(X, Wt, adj, C);
  gibbs_kernel<<<1, 32, 0, stream>>>(C, ES, Sbits);
}
